// TemporalAttention_10771777978969
// MI455X (gfx1250) — matrix operand order repaired and checked by reading, not compiled, not run on hardware
//
#include <hip/hip_runtime.h>
#include <hip/hip_bf16.h>
#include <math.h>

// ---------------- types / helpers ----------------
typedef __attribute__((ext_vector_type(16))) __bf16 vbf16;
typedef __attribute__((ext_vector_type(8)))  float  v8f;

static __device__ __forceinline__ unsigned short f2bf(float f) {
  unsigned u = __builtin_bit_cast(unsigned, f);
  u += 0x7fffu + ((u >> 16) & 1u);          // round-to-nearest-even
  return (unsigned short)(u >> 16);
}
static __device__ __forceinline__ float bf2f(unsigned short h) {
  unsigned u = ((unsigned)h) << 16;
  return __builtin_bit_cast(float, u);
}

union Frag {
  vbf16 v;
  unsigned short us[16];
  uint4 q[2];
};

// A fragment (16x32 bf16, MxK): row-major source, lane m=lane&15 is the row,
// half=lane>>4 selects K pattern {half*8..+7, 16+half*8..+7}.
static __device__ __forceinline__ vbf16 loadA(const unsigned short* p, int rowStride,
                                              int row, int k0, int lane) {
  int half = lane >> 4;
  const unsigned short* r = p + (size_t)row * rowStride + k0 + half * 8;
  Frag f;
  f.q[0] = *reinterpret_cast<const uint4*>(r);
  f.q[1] = *reinterpret_cast<const uint4*>(r + 16);
  return f.v;
}
// B fragment (32x16 bf16, KxN) from column-contiguous storage Bt (N rows x K):
// lane n=lane&15 is the column; half=lane>>4 selects K pattern {half*8..+7, 16+half*8..+7}.
static __device__ __forceinline__ vbf16 loadB(const unsigned short* p, int rowStride,
                                              int col, int k0, int lane) {
  int half = lane >> 4;
  const unsigned short* r = p + (size_t)col * rowStride + k0 + half * 8;
  Frag f;
  f.q[0] = *reinterpret_cast<const uint4*>(r);
  f.q[1] = *reinterpret_cast<const uint4*>(r + 16);
  return f.v;
}
static __device__ __forceinline__ v8f wmma_bf16(vbf16 a, vbf16 b, v8f c) {
  return __builtin_amdgcn_wmma_f32_16x16x32_bf16(false, a, false, b, (short)0, c,
                                                 false, false);
}

// ---------------- constants ----------------
#define NBL   2048      // B*L rows
#define CR    2048      // C*R
#define HD    512       // H*DQK
#define QKW   576       // HD + DQK  (fused q|k output width)
#define CV    2048      // C*RV
#define LQ    1024      // L

// ---------------- prep ----------------
__global__ void k_cvt_x(const float* x, unsigned short* xb, int n) {
  int i = blockIdx.x * blockDim.x + threadIdx.x;
  if (i < n) xb[i] = f2bf(x[i]);
}

__global__ void k_prep_w(const float* Wq, const float* Wk, const float* Wv,
                         const float* Wo, unsigned short* wqkt,
                         unsigned short* wvt, unsigned short* wot) {
  int i = blockIdx.x * blockDim.x + threadIdx.x;
  if (i < QKW * CR) {                       // Wqkt[n][k] = [Wq|Wk]^T
    int n = i >> 11, k = i & 2047;
    float v = (n < HD) ? Wq[(size_t)k * HD + n] : Wk[(size_t)k * 64 + (n - HD)];
    wqkt[i] = f2bf(v);
  } else if (i < QKW * CR + 65536) {        // Wvt[c][o][r] = Wv[c][r][o]
    int j = i - QKW * CR;
    int c = j >> 10, o = (j >> 5) & 31, r = j & 31;
    wvt[j] = f2bf(Wv[(c * 32 + r) * 32 + o]);
  } else if (i < QKW * CR + 131072) {       // Wot[c][r][v] = Wo[c][v][r]
    int j = i - (QKW * CR + 65536);
    int c = j >> 10, r = (j >> 5) & 31, vv = j & 31;
    wot[j] = f2bf(Wo[(c * 32 + vv) * 32 + r]);
  }
}

// ---------------- fused q|k projection: (2048 x 2048) @ (2048 x 576) ----------------
__global__ void k_qk(const unsigned short* xb, const unsigned short* wqkt,
                     const float* bq, const float* bk, unsigned short* qk) {
  int wave = threadIdx.x >> 5, lane = threadIdx.x & 31;
  int t = blockIdx.x * (blockDim.x >> 5) + wave;
  if (t >= 128 * 36) return;                 // wave-uniform guard
  int tM = t % 128, tN = t / 128;
  int mrow = lane & 15, half = lane >> 4;
  v8f acc{};
  for (int k0 = 0; k0 < CR; k0 += 32) {
    if (k0 + 32 < CR)
      __builtin_prefetch(xb + (size_t)(tM * 16 + mrow) * CR + k0 + 32, 0, 1);
    vbf16 a = loadA(xb, CR, tM * 16 + mrow, k0, lane);
    vbf16 b = loadB(wqkt, CR, tN * 16 + mrow, k0, lane);
    acc = wmma_bf16(a, b, acc);
  }
  int col = tN * 16 + mrow;
  float bias = (col < HD) ? bq[col] : bk[col - HD];
  for (int g = 0; g < 8; ++g) {
    int row = tM * 16 + g + half * 8;
    qk[(size_t)row * QKW + col] = f2bf(acc[g] + bias);
  }
}

// ---------------- per-channel value projection -> v^T (b, cv, l) ----------------
__global__ void k_vproj(const unsigned short* xb, const unsigned short* wvt,
                        const float* bv, unsigned short* vt) {
  int wave = threadIdx.x >> 5, lane = threadIdx.x & 31;
  int t = blockIdx.x * (blockDim.x >> 5) + wave;
  if (t >= 128 * 64 * 2) return;
  int nt = t & 1, c = (t >> 1) & 63, tM = t >> 7;
  int mrow = lane & 15, half = lane >> 4;
  vbf16 a = loadA(xb, CR, tM * 16 + mrow, c * 32, lane);
  vbf16 b = loadB(wvt, 32, c * 32 + nt * 16 + mrow, 0, lane);
  v8f acc{};
  acc = wmma_bf16(a, b, acc);
  int o = nt * 16 + mrow;
  float bias = bv[c * 32 + o];
  int cv = c * 32 + o;
  int blbase = tM * 16;
  int bidx = blbase >> 10;
  int lbase = (blbase & 1023) + half * 8;
  unsigned short us[8];
  for (int g = 0; g < 8; ++g) us[g] = f2bf(acc[g] + bias);
  uint4 pk;
  pk.x = us[0] | ((unsigned)us[1] << 16);
  pk.y = us[2] | ((unsigned)us[3] << 16);
  pk.z = us[4] | ((unsigned)us[5] << 16);
  pk.w = us[6] | ((unsigned)us[7] << 16);
  *reinterpret_cast<uint4*>(vt + ((size_t)(bidx * CV + cv)) * LQ + lbase) = pk;
}

// ---------------- attention: logits + fp32 softmax + head-sum ----------------
// one workgroup per (b, 16-row l tile); wave w owns s in [w*128, w*128+128)
__global__ void k_attn(const unsigned short* qk, unsigned short* asum) {
  __shared__ float red[8][16];
  int blk = blockIdx.x;                     // 128 blocks
  int b = blk >> 6, lt = blk & 63;
  int wave = threadIdx.x >> 5, lane = threadIdx.x & 31;
  int mrow = lane & 15, half = lane >> 4;
  int rowbase = b * 1024 + lt * 16;
  v8f asacc[8];
  for (int j = 0; j < 8; ++j) asacc[j] = v8f{};
  for (int h = 0; h < 8; ++h) {
    vbf16 qa0 = loadA(qk, QKW, rowbase + mrow, h * 64, lane);
    vbf16 qa1 = loadA(qk, QKW, rowbase + mrow, h * 64 + 32, lane);
    v8f lg[8];
    for (int j = 0; j < 8; ++j) {
      int st = wave * 8 + j;
      vbf16 kb0 = loadB(qk, QKW, b * 1024 + st * 16 + mrow, HD, lane);
      vbf16 kb1 = loadB(qk, QKW, b * 1024 + st * 16 + mrow, HD + 32, lane);
      v8f acc{};
      acc = wmma_bf16(qa0, kb0, acc);
      acc = wmma_bf16(qa1, kb1, acc);
      lg[j] = acc * 0.125f;                  // 1/sqrt(64)
    }
    // row max: across this wave's 8 tiles, then 16 lanes, then 8 waves via LDS
    for (int g = 0; g < 8; ++g) {
      float m = lg[0][g];
      for (int j = 1; j < 8; ++j) m = fmaxf(m, lg[j][g]);
      for (int off = 1; off < 16; off <<= 1) m = fmaxf(m, __shfl_xor(m, off, 16));
      if (mrow == 0) red[wave][half * 8 + g] = m;
    }
    __syncthreads();
    float rmax[8];
    for (int g = 0; g < 8; ++g) {
      float m = red[0][half * 8 + g];
      for (int w = 1; w < 8; ++w) m = fmaxf(m, red[w][half * 8 + g]);
      rmax[g] = m;
    }
    __syncthreads();
    // exp + row sum
    for (int g = 0; g < 8; ++g) {
      float s = 0.f;
      for (int j = 0; j < 8; ++j) {
        float e = __expf(lg[j][g] - rmax[g]);
        lg[j][g] = e;
        s += e;
      }
      for (int off = 1; off < 16; off <<= 1) s += __shfl_xor(s, off, 16);
      if (mrow == 0) red[wave][half * 8 + g] = s;
    }
    __syncthreads();
    for (int g = 0; g < 8; ++g) {
      float s = 0.f;
      for (int w = 0; w < 8; ++w) s += red[w][half * 8 + g];
      float iv = 1.0f / s;
      for (int j = 0; j < 8; ++j) asacc[j][g] += lg[j][g] * iv;
    }
    __syncthreads();
  }
  for (int j = 0; j < 8; ++j) {
    int s = wave * 128 + j * 16 + mrow;
    for (int g = 0; g < 8; ++g) {
      int l = lt * 16 + g + half * 8;
      asum[((size_t)(b * 1024 + l)) * LQ + s] = f2bf(asacc[j][g]);
    }
  }
}

// ---------------- AV: attn_sum (bl x 1024) @ v^T -> ymid (bl x 2048), x 1/H ----------------
__global__ void k_av(const unsigned short* asum, const unsigned short* vt,
                     unsigned short* ymid) {
  int wave = threadIdx.x >> 5, lane = threadIdx.x & 31;
  int t = blockIdx.x * (blockDim.x >> 5) + wave;
  if (t >= 128 * 128) return;
  int tN = t & 127, tM = t >> 7;
  int mrow = lane & 15, half = lane >> 4;
  int b = (tM * 16) >> 10;
  v8f acc{};
  for (int k0 = 0; k0 < LQ; k0 += 32) {
    if (k0 + 32 < LQ)
      __builtin_prefetch(asum + (size_t)(tM * 16 + mrow) * LQ + k0 + 32, 0, 1);
    vbf16 a = loadA(asum, LQ, tM * 16 + mrow, k0, lane);
    vbf16 bf = loadB(vt, LQ, b * CV + tN * 16 + mrow, k0, lane);
    acc = wmma_bf16(a, bf, acc);
  }
  int cv = tN * 16 + mrow;
  for (int g = 0; g < 8; ++g) {
    int row = tM * 16 + g + half * 8;
    ymid[(size_t)row * CV + cv] = f2bf(acc[g] * 0.125f);   // 1/H
  }
}

// ---------------- sink-query branch (tiny, fp32 VALU) ----------------
__global__ void k_sink(const unsigned short* qk, const float* sink_q, float* wsink) {
  __shared__ float wl[1024];
  __shared__ float red[256];
  int b = blockIdx.x, tid = threadIdx.x;
  for (int i = tid; i < 1024; i += 256) wl[i] = 0.f;
  __syncthreads();
  for (int rw = 0; rw < 32; ++rw) {          // H*NSINK rows
    const float* sq = sink_q + rw * 64;
    float lg[4];
    for (int j = 0; j < 4; ++j) {
      int l = tid + j * 256;
      const unsigned short* kp = qk + ((size_t)(b * 1024 + l)) * QKW + HD;
      float s = 0.f;
      for (int d = 0; d < 64; ++d) s += sq[d] * bf2f(kp[d]);
      lg[j] = s * 0.125f;
    }
    float m = fmaxf(fmaxf(lg[0], lg[1]), fmaxf(lg[2], lg[3]));
    red[tid] = m; __syncthreads();
    for (int s = 128; s > 0; s >>= 1) {
      if (tid < s) red[tid] = fmaxf(red[tid], red[tid + s]);
      __syncthreads();
    }
    m = red[0]; __syncthreads();
    float sm = 0.f;
    for (int j = 0; j < 4; ++j) { lg[j] = __expf(lg[j] - m); sm += lg[j]; }
    red[tid] = sm; __syncthreads();
    for (int s = 128; s > 0; s >>= 1) {
      if (tid < s) red[tid] += red[tid + s];
      __syncthreads();
    }
    float inv = 1.0f / red[0]; __syncthreads();
    for (int j = 0; j < 4; ++j) wl[tid + j * 256] += lg[j] * inv;
  }
  __syncthreads();
  for (int i = tid; i < 1024; i += 256)
    wsink[b * 1024 + i] = wl[i] * (1.0f / 32.0f);   // 1/(H*NSINK)
}

__global__ void k_ysink(const unsigned short* vt, const float* wsink, float* ysink) {
  int g = blockIdx.x * blockDim.x + threadIdx.x;
  if (g >= 2 * CV) return;
  int b = g >> 11, cv = g & 2047;
  const unsigned short* vp = vt + ((size_t)(b * CV + cv)) * LQ;
  const float* wp = wsink + b * 1024;
  float s = 0.f;
  for (int l = 0; l < 1024; ++l) s += bf2f(vp[l]) * wp[l];
  ysink[g] = s;
}

// ---------------- output projection + bias + residual ----------------
__global__ void k_out(const unsigned short* ymid, const float* ysink,
                      const unsigned short* wot, const float* bo,
                      const float* x, float* out) {
  int wave = threadIdx.x >> 5, lane = threadIdx.x & 31;
  int t = blockIdx.x * (blockDim.x >> 5) + wave;
  if (t >= 128 * 64 * 2) return;
  int nt = t & 1, c = (t >> 1) & 63, tM = t >> 7;
  int mrow = lane & 15, half = lane >> 4;
  int b = (tM * 16) >> 10;
  int row = tM * 16 + mrow;
  Frag fa;                                   // A = ymid + y_sink broadcast
  for (int e = 0; e < 16; ++e) {
    int k = (e < 8) ? (half * 8 + e) : (16 + half * 8 + (e - 8));
    int cvk = c * 32 + k;
    float yv = bf2f(ymid[(size_t)row * CV + cvk]) + ysink[b * CV + cvk];
    fa.us[e] = f2bf(yv);
  }
  vbf16 bf = loadB(wot, 32, c * 32 + nt * 16 + mrow, 0, lane);
  v8f acc{};
  acc = wmma_bf16(fa.v, bf, acc);
  int r = nt * 16 + mrow;
  float bias = bo[c * 32 + r];
  for (int g = 0; g < 8; ++g) {
    int rr = tM * 16 + g + half * 8;
    size_t idx = ((size_t)rr * 64 + c) * 32 + r;
    out[idx] = acc[g] + bias + x[idx];
  }
}

// ---------------- launcher ----------------
extern "C" void kernel_launch(void* const* d_in, const int* in_sizes, int n_in,
                              void* d_out, int out_size, void* d_ws, size_t ws_size,
                              hipStream_t stream) {
  (void)in_sizes; (void)n_in; (void)out_size; (void)ws_size;
  const float* x  = (const float*)d_in[0];
  const float* Wq = (const float*)d_in[1];
  const float* bq = (const float*)d_in[2];
  const float* Wk = (const float*)d_in[3];
  const float* bk = (const float*)d_in[4];
  const float* Wv = (const float*)d_in[5];
  const float* bv = (const float*)d_in[6];
  const float* Wo = (const float*)d_in[7];
  const float* bo = (const float*)d_in[8];
  const float* sq = (const float*)d_in[9];
  float* out = (float*)d_out;

  char* ws = (char*)d_ws;
  unsigned short* xb   = (unsigned short*)(ws + 0);         // 8 MB  x bf16
  unsigned short* qk   = (unsigned short*)(ws + 8388608);   // 2.25MB fused q|k
  unsigned short* vt   = (unsigned short*)(ws + 10747904);  // 8 MB  v^T
  unsigned short* asum = (unsigned short*)(ws + 19136512);  // 4 MB  head-summed attn
  unsigned short* ymid = (unsigned short*)(ws + 23330816);  // 8 MB  AV result
  unsigned short* wqkt = (unsigned short*)(ws + 31719424);  // 2.25MB [Wq|Wk]^T
  unsigned short* wvt  = (unsigned short*)(ws + 34078720);  // 128KB Wv^T
  unsigned short* wot  = (unsigned short*)(ws + 34209792);  // 128KB Wo^T
  float* wsink = (float*)(ws + 34340864);                   // 8KB
  float* ysink = (float*)(ws + 34349056);                   // 16KB

  k_cvt_x <<<(4194304 + 255) / 256, 256, 0, stream>>>(x, xb, 4194304);
  k_prep_w<<<(1310720 + 255) / 256, 256, 0, stream>>>(Wq, Wk, Wv, Wo, wqkt, wvt, wot);
  k_qk    <<<576, 256, 0, stream>>>(xb, wqkt, bq, bk, qk);
  k_vproj <<<2048, 256, 0, stream>>>(xb, wvt, bv, vt);
  k_attn  <<<128, 256, 0, stream>>>(qk, asum);
  k_av    <<<2048, 256, 0, stream>>>(asum, vt, ymid);
  k_sink  <<<2, 256, 0, stream>>>(qk, sq, wsink);
  k_ysink <<<16, 256, 0, stream>>>(vt, wsink, ysink);
  k_out   <<<2048, 256, 0, stream>>>(ymid, ysink, wot, bo, x, out);
}
